// mlpmixer_31550829756687
// MI455X (gfx1250) — matrix operand order repaired and checked by reading, not compiled, not run on hardware
//
#include <hip/hip_runtime.h>
#include <hip/hip_bf16.h>

typedef __attribute__((ext_vector_type(16))) _Float16 v16h;
typedef __attribute__((ext_vector_type(8)))  _Float16 v8h;
typedef __attribute__((ext_vector_type(8)))  float    v8f;

constexpr int L_    = 8;     // layers
constexpr int REP_  = 2;     // repeats per layer
constexpr int DIM_  = 256;   // mixer hidden
constexpr int DIM2_ = 128;   // head dim
constexpr int SPB   = 8;     // samples (waves) per block
constexpr int HSS   = 40;    // h row stride (floats, 160B -> 32B-aligned v8f slots)
constexpr int ASS   = 40;    // A/T row stride (halfs, 80B -> 16B-aligned v8h slots)
constexpr int WPL   = 4 * DIM_ * 32;      // 32768 halfs of weights per layer
constexpr int FC1_OFF   = L_ * WPL;       // fc1 f16 weights offset in ws (halfs)
constexpr int CVT_TOTAL = 4 * L_ * DIM_ * 32 + DIM2_ * 32;  // 266240 elements

constexpr size_t SMEM_BYTES =
    (size_t)SPB * 32 * HSS * 4      // h state, f32            40960
  + (size_t)WPL * 2                 // staged layer weights    65536
  + (size_t)SPB * 32 * ASS * 2      // A tiles, f16            20480
  + (size_t)SPB * 32 * ASS * 2      // T scratch tiles, f16    20480
  + (size_t)SPB * 128 * 4;          // pooled features, f32     4096  => 151552 B

// ---------------- weight conversion: f32 -> f16 into workspace ----------------
// ws layout (halfs): [layer]{ w11[256*32] | w12[32*256] | w21[256*32] | w22[32*256] } then fc1[128*32]
__global__ void cvt_weights_k(const float* __restrict__ w11, const float* __restrict__ w12,
                              const float* __restrict__ w21, const float* __restrict__ w22,
                              const float* __restrict__ fc1w, _Float16* __restrict__ ws) {
  int idx = blockIdx.x * blockDim.x + threadIdx.x;
  const int PER = L_ * DIM_ * 32;  // 65536
  const int BLK = DIM_ * 32;       // 8192
  if (idx < PER) {
    int l = idx / BLK, r = idx % BLK;
    ws[l * WPL + r] = (_Float16)w11[idx];
  } else if (idx < 2 * PER) {
    int i2 = idx - PER; int l = i2 / BLK, r = i2 % BLK;
    ws[l * WPL + BLK + r] = (_Float16)w12[i2];
  } else if (idx < 3 * PER) {
    int i2 = idx - 2 * PER; int l = i2 / BLK, r = i2 % BLK;
    ws[l * WPL + 2 * BLK + r] = (_Float16)w21[i2];
  } else if (idx < 4 * PER) {
    int i2 = idx - 3 * PER; int l = i2 / BLK, r = i2 % BLK;
    ws[l * WPL + 3 * BLK + r] = (_Float16)w22[i2];
  } else if (idx < CVT_TOTAL) {
    ws[FC1_OFF + (idx - 4 * PER)] = (_Float16)fc1w[idx - 4 * PER];
  }
}

// ---------------- helpers ----------------
__device__ __forceinline__ float hswish(float x) {
  float t = fminf(fmaxf(x + 3.0f, 0.0f), 6.0f);
  return x * t * (1.0f / 6.0f);
}

// A-matrix fragment, 16x32 f16 (ISA 7.12.2): lane<16 holds K 0..7 & 16..23, lane>=16 holds K 8..15 & 24..31
__device__ __forceinline__ v16h loadFragA(const _Float16* base, int stride, int rowBase,
                                          int kBase, int lane) {
  int l = lane & 15, hi = lane >> 4;
  const _Float16* p = base + (rowBase + l) * stride + kBase + hi * 8;
  v8h a = *(const v8h*)p;
  v8h b = *(const v8h*)(p + 16);
  v16h r;
#pragma unroll
  for (int i = 0; i < 8; ++i) { r[i] = a[i]; r[i + 8] = b[i]; }
  return r;
}

// B-matrix fragment, 32x16 f16: lane<16 holds K 0..7 & 16..23, lane>=16 holds K 8..15 & 24..31 (N = lane&15)
// Source must be N-major x K row-major (PyTorch [out,in] layout).
__device__ __forceinline__ v16h loadFragB(const _Float16* base, int stride, int rowBase,
                                          int kBase, int lane) {
  int l = lane & 15, hi = lane >> 4;
  const _Float16* p = base + (rowBase + l) * stride + kBase + hi * 8;
  v8h a = *(const v8h*)p;
  v8h b = *(const v8h*)(p + 16);
  v16h r;
#pragma unroll
  for (int i = 0; i < 8; ++i) { r[i] = a[i]; r[i + 8] = b[i]; }
  return r;
}

__device__ __forceinline__ v8f wmma_f32(v16h a, v16h b, v8f c) {
  return __builtin_amdgcn_wmma_f32_16x16x32_f16(false, a, false, b, (short)0, c, false, false);
}

// One mixing branch. GEMM1 is computed transposed (A-frag = W1 rows, B-frag = activations)
// so the hidden dim lands on the VGPR axis: bias rides in as the WMMA C operand and the
// hardswished f16 tile is stored with ONE ds_store_b128 per tile. GEMM2 uses the
// formulation whose output has row=lane-axis / col=VGPR-axis so the residual update is
// contiguous v8f loads/stores seeded through the WMMA C chain.
template <bool TOKEN>
__device__ void mix_branch(float* hS, _Float16* aS, _Float16* tS,
                           const _Float16* W1, const _Float16* W2,
                           const float* lnG, const float* lnB,
                           const float* bias1, const float* bias2, int lane) {
  int l15 = lane & 15, hi8 = (lane >> 4) * 8;

  // LayerNorm over last dim: lane = row i of h
  const float* hr = hS + lane * HSS;
  float sum = 0.f, ss = 0.f;
#pragma unroll
  for (int j = 0; j < 32; ++j) { float v = hr[j]; sum += v; ss += v * v; }
  float m   = sum * (1.f / 32.f);
  float var = ss * (1.f / 32.f) - m * m;
  float rs  = rsqrtf(var + 1e-5f);
#pragma unroll
  for (int j = 0; j < 32; ++j) {
    float nv = (hr[j] - m) * rs * lnG[j] + lnB[j];
    if (TOKEN) aS[j * ASS + lane] = (_Float16)nv;   // store LN(h)^T for token mixing
    else       aS[lane * ASS + j] = (_Float16)nv;
  }

  // activation B-fragments (constant across chunks): aS is [m][k] row-major
  v16h ba0 = loadFragB(aS, ASS, 0, 0, lane);
  v16h ba1 = loadFragB(aS, ASS, 16, 0, lane);

  // residual accumulators u[vt][lt], seeded with h + bias2.
  // element (v): h row = 16*lt + l15, col = 16*vt + hi8 + v
  v8f u00, u01, u10, u11;
  {
    const float* h00 = hS + l15 * HSS + hi8;
    const float* h10 = hS + (16 + l15) * HSS + hi8;
    u00 = *(const v8f*)(h00);
    u01 = *(const v8f*)(h10);
    u10 = *(const v8f*)(h00 + 16);
    u11 = *(const v8f*)(h10 + 16);
    if (TOKEN) {  // bias2 indexed by lane axis
      float b0 = bias2[l15], b1 = bias2[16 + l15];
#pragma unroll
      for (int v = 0; v < 8; ++v) { u00[v] += b0; u01[v] += b1; u10[v] += b0; u11[v] += b1; }
    } else {      // bias2 indexed by VGPR axis
      v8f c0 = *(const v8f*)(bias2 + hi8);
      v8f c1 = *(const v8f*)(bias2 + 16 + hi8);
      u00 += c0; u01 += c0; u10 += c1; u11 += c1;
    }
  }

  for (int c = 0; c < DIM_ / 32; ++c) {
    // ---- GEMM1 (transposed): D'[d][m] = W1_chunk x A^T, bias1 injected via C ----
    v16h wa0 = loadFragA(W1, 32, 32 * c, 0, lane);
    v16h wa1 = loadFragA(W1, 32, 32 * c + 16, 0, lane);
    v8f bc0 = *(const v8f*)(bias1 + 32 * c + hi8);        // d-tile 0
    v8f bc1 = *(const v8f*)(bias1 + 32 * c + 16 + hi8);   // d-tile 1
    v8f d00 = wmma_f32(wa0, ba0, bc0);   // (dt=0, mt=0)
    v8f d01 = wmma_f32(wa0, ba1, bc0);   // (dt=0, mt=1)
    v8f d10 = wmma_f32(wa1, ba0, bc1);   // (dt=1, mt=0)
    v8f d11 = wmma_f32(wa1, ba1, bc1);   // (dt=1, mt=1)

    // hardswish -> f16, one b128 store per tile: tS[m][dlocal], dlocal = 16*dt + hi8 + v
    {
      v8h t0, t1, t2, t3;
#pragma unroll
      for (int v = 0; v < 8; ++v) {
        t0[v] = (_Float16)hswish(d00[v]);
        t1[v] = (_Float16)hswish(d01[v]);
        t2[v] = (_Float16)hswish(d10[v]);
        t3[v] = (_Float16)hswish(d11[v]);
      }
      *(v8h*)(tS + l15 * ASS + hi8)             = t0;
      *(v8h*)(tS + (16 + l15) * ASS + hi8)      = t1;
      *(v8h*)(tS + l15 * ASS + 16 + hi8)        = t2;
      *(v8h*)(tS + (16 + l15) * ASS + 16 + hi8) = t3;
    }

    // ---- GEMM2 accumulate into u (in-wave DS order makes the tS bounce barrier-free) ----
    if (TOKEN) {  // U[j][i'] = T x W2^T : A = T (tS), B = w12
      v16h ta0 = loadFragA(tS, ASS, 0, 0, lane);
      v16h ta1 = loadFragA(tS, ASS, 16, 0, lane);
      v16h wb0 = loadFragB(W2, DIM_, 0, 32 * c, lane);
      v16h wb1 = loadFragB(W2, DIM_, 16, 32 * c, lane);
      u00 = wmma_f32(ta0, wb0, u00);   // vt = j-tile 0, lt = i'-tile 0
      u01 = wmma_f32(ta0, wb1, u01);
      u10 = wmma_f32(ta1, wb0, u10);
      u11 = wmma_f32(ta1, wb1, u11);
    } else {      // U^T[i2][m] = W2 x T^T : A = w22, B = T (tS)
      v16h wa2 = loadFragA(W2, DIM_, 0, 32 * c, lane);
      v16h wa3 = loadFragA(W2, DIM_, 16, 32 * c, lane);
      v16h tb0 = loadFragB(tS, ASS, 0, 0, lane);
      v16h tb1 = loadFragB(tS, ASS, 16, 0, lane);
      u00 = wmma_f32(wa2, tb0, u00);   // vt = i2-tile 0, lt = m-tile 0
      u01 = wmma_f32(wa2, tb1, u01);
      u10 = wmma_f32(wa3, tb0, u10);
      u11 = wmma_f32(wa3, tb1, u11);
    }
  }

  // residual writeback: contiguous v8f stores
  {
    float* h00 = hS + l15 * HSS + hi8;
    float* h10 = hS + (16 + l15) * HSS + hi8;
    *(v8f*)(h00)      = u00;
    *(v8f*)(h10)      = u01;
    *(v8f*)(h00 + 16) = u10;
    *(v8f*)(h10 + 16) = u11;
  }
}

// ---------------- fused mixer kernel: one wave per sample ----------------
// LDS (151.5 KB) allows 2 blocks / WGP = 4 waves per SIMD32; request exactly that so
// the register allocator stays within file/4 (<=256 VGPRs, no vgpr-msb switching).
__global__ __launch_bounds__(SPB * 32, 4)
void mixer_main_k(const float* __restrict__ x, const float* __restrict__ conv_w,
                  const float* __restrict__ conv_b, const float* __restrict__ fc0_w,
                  const float* __restrict__ fc0_b, const float* __restrict__ ln1_g,
                  const float* __restrict__ ln1_b, const float* __restrict__ b11,
                  const float* __restrict__ b12, const float* __restrict__ ln2_g,
                  const float* __restrict__ ln2_b, const float* __restrict__ b21,
                  const float* __restrict__ b22, const _Float16* __restrict__ wsH,
                  const float* __restrict__ fc1_b, const float* __restrict__ fcout_w,
                  const float* __restrict__ fcout_b, float* __restrict__ out, int nB) {
  extern __shared__ char smem[];
  float*    hAll = (float*)smem;
  _Float16* sW   = (_Float16*)(smem + (size_t)SPB * 32 * HSS * 4);
  _Float16* aAll = sW + WPL;
  _Float16* tAll = aAll + SPB * 32 * ASS;
  float*    pAll = (float*)(tAll + SPB * 32 * ASS);

  int wave = threadIdx.x >> 5, lane = threadIdx.x & 31;
  int sample = blockIdx.x * SPB + wave;
  bool valid = sample < nB;                  // uniform per wave -> EXEC all-ones in WMMA

  float*    hS = hAll + wave * 32 * HSS;
  _Float16* aS = aAll + wave * 32 * ASS;
  _Float16* tS = tAll + wave * 32 * ASS;
  float*    pS = pAll + wave * 128;
  int l15 = lane & 15, hiRow = (lane >> 4) * 8;

  // ---- prologue: h = (1x1 conv over 3 chans) @ fc0^T + fc0_b ----
  if (valid) {
    float cw0 = conv_w[0], cw1 = conv_w[1], cw2 = conv_w[2], cb = conv_b[0];
    const float* xb = x + (size_t)sample * 3 * 1024;
#pragma unroll 4
    for (int i = 0; i < 32; ++i) {
      float v = xb[i * 32 + lane] * cw0 + xb[1024 + i * 32 + lane] * cw1 +
                xb[2048 + i * 32 + lane] * cw2 + cb;
      hS[i * HSS + lane] = v;
    }
    float f0[32];
#pragma unroll
    for (int k = 0; k < 32; ++k) f0[k] = fc0_w[lane * 32 + k];
    float f0b = fc0_b[lane];
    for (int i = 0; i < 32; ++i) {
      float acc = f0b;
#pragma unroll
      for (int k = 0; k < 32; ++k) acc += hS[i * HSS + k] * f0[k];
      hS[i * HSS + lane] = acc;   // reads of row i precede this store (in-wave DS order)
    }
  }

  // ---- mixer layers; weights staged into LDS once per layer ----
  for (int layer = 0; layer < L_; ++layer) {
    __syncthreads();
    {
      const uint4* src = (const uint4*)(wsH + (size_t)layer * WPL);
      uint4* dst = (uint4*)sW;
      for (int t = threadIdx.x; t < WPL / 8; t += SPB * 32) dst[t] = src[t];
    }
    __syncthreads();
    if (valid) {
      const _Float16* w11h = sW;
      const _Float16* w12h = sW + DIM_ * 32;
      const _Float16* w21h = sW + 2 * DIM_ * 32;
      const _Float16* w22h = sW + 3 * DIM_ * 32;
      for (int r = 0; r < REP_; ++r) {
        mix_branch<true >(hS, aS, tS, w11h, w12h, ln1_g + layer * 32, ln1_b + layer * 32,
                          b11 + layer * DIM_, b12 + layer * 32, lane);
        mix_branch<false>(hS, aS, tS, w21h, w22h, ln2_g + layer * 32, ln2_b + layer * 32,
                          b21 + layer * DIM_, b22 + layer * 32, lane);
      }
    }
  }

  // ---- epilogue: fc1 + hardswish + AvgPool1d(32) + fcout ----
  if (valid) {
#pragma unroll
    for (int j = 0; j < 32; ++j) aS[lane * ASS + j] = (_Float16)hS[lane * HSS + j];
    // transposed production again: Y^T[d][i] = fc1_chunk x h^T, bias via C operand.
    v16h hb0 = loadFragB(aS, ASS, 0, 0, lane);   // h as B-frags, [i][j] row-major
    v16h hb1 = loadFragB(aS, ASS, 16, 0, lane);
    const _Float16* fc1h = wsH + FC1_OFF;        // [128][32] f16, N-major (L2 resident)
    for (int c = 0; c < DIM2_ / 32; ++c) {
      v16h fa0 = loadFragA(fc1h, 32, 32 * c, 0, lane);
      v16h fa1 = loadFragA(fc1h, 32, 32 * c + 16, 0, lane);
      v8f bc0 = *(const v8f*)(fc1_b + 32 * c + hiRow);
      v8f bc1 = *(const v8f*)(fc1_b + 32 * c + 16 + hiRow);
      v8f d00 = wmma_f32(fa0, hb0, bc0);   // lane = i-axis, VGPR = d-axis
      v8f d01 = wmma_f32(fa0, hb1, bc0);
      v8f d10 = wmma_f32(fa1, hb0, bc1);
      v8f d11 = wmma_f32(fa1, hb1, bc1);
      // pool: mean over the 32 d's of this chunk (= AvgPool group c).
      // lane pair (l, l+16) holds complementary d-halves of the same i.
      float s0 = 0.f, s1 = 0.f;
#pragma unroll
      for (int v = 0; v < 8; ++v) {
        s0 += hswish(d00[v]) + hswish(d10[v]);   // i = l15
        s1 += hswish(d01[v]) + hswish(d11[v]);   // i = 16 + l15
      }
      s0 += __shfl_xor(s0, 16, 32);
      s1 += __shfl_xor(s1, 16, 32);
      if (hiRow == 0) {                           // lanes 0..15 write P[i][c]
        pS[l15 * 4 + c]        = s0 * (1.f / 32.f);
        pS[(16 + l15) * 4 + c] = s1 * (1.f / 32.f);
      }
    }
    if (lane < 10) {
      float acc = fcout_b[lane];
      const float* wr = fcout_w + lane * DIM2_;
#pragma unroll 8
      for (int k = 0; k < DIM2_; ++k) acc += pS[k] * wr[k];
      out[(size_t)sample * 10 + lane] = acc;
    }
  }
}

// ---------------- host entry ----------------
extern "C" void kernel_launch(void* const* d_in, const int* in_sizes, int n_in,
                              void* d_out, int out_size, void* d_ws, size_t ws_size,
                              hipStream_t stream) {
  (void)n_in; (void)out_size; (void)ws_size;
  const float* x      = (const float*)d_in[0];
  const float* conv_w = (const float*)d_in[1];
  const float* conv_b = (const float*)d_in[2];
  const float* fc0_w  = (const float*)d_in[3];
  const float* fc0_b  = (const float*)d_in[4];
  const float* ln1_g  = (const float*)d_in[5];
  const float* ln1_b  = (const float*)d_in[6];
  const float* w11    = (const float*)d_in[7];
  const float* b11    = (const float*)d_in[8];
  const float* w12    = (const float*)d_in[9];
  const float* b12    = (const float*)d_in[10];
  const float* ln2_g  = (const float*)d_in[11];
  const float* ln2_b  = (const float*)d_in[12];
  const float* w21    = (const float*)d_in[13];
  const float* b21    = (const float*)d_in[14];
  const float* w22    = (const float*)d_in[15];
  const float* b22    = (const float*)d_in[16];
  const float* fc1_w  = (const float*)d_in[17];
  const float* fc1_b  = (const float*)d_in[18];
  const float* fcout_w = (const float*)d_in[19];
  const float* fcout_b = (const float*)d_in[20];

  int nB = in_sizes[0] / (3 * 32 * 32);
  _Float16* wsH = (_Float16*)d_ws;  // needs 266240*2 = 532480 bytes of scratch

  cvt_weights_k<<<(CVT_TOTAL + 255) / 256, 256, 0, stream>>>(w11, w12, w21, w22, fc1_w, wsH);

  // idempotent, non-stream call; safe under graph capture
  (void)hipFuncSetAttribute((const void*)mixer_main_k,
                            hipFuncAttributeMaxDynamicSharedMemorySize, (int)SMEM_BYTES);

  int blocks = (nB + SPB - 1) / SPB;
  mixer_main_k<<<blocks, SPB * 32, SMEM_BYTES, stream>>>(
      x, conv_w, conv_b, fc0_w, fc0_b, ln1_g, ln1_b, b11, b12, ln2_g, ln2_b,
      b21, b22, wsH, fc1_b, fcout_w, fcout_b, (float*)d_out, nB);
}
